// SlowFastFusionArc1_19464791785861
// MI455X (gfx1250) — matrix operand order repaired and checked by reading, not compiled, not run on hardware
//
#include <hip/hip_runtime.h>
#include <math.h>

// ---------------- types ----------------
typedef __bf16 bf16_t;
typedef bf16_t bf16x4 __attribute__((ext_vector_type(4)));
typedef bf16_t bf16x8 __attribute__((ext_vector_type(8)));
typedef bf16_t v16bf  __attribute__((ext_vector_type(16)));
typedef float  v8f    __attribute__((ext_vector_type(8)));
typedef int    v4i    __attribute__((ext_vector_type(4)));

union Frag { v16bf v; bf16x8 h[2]; };

// gfx1250 async global->LDS copy (ASYNCcnt-tracked), guarded per-builtin.
// Signature (from compiler diagnostic / LLVM def "vV4i*1V4i*3IiIi"):
//   void __builtin_amdgcn_global_load_async_to_lds_b128(
//       v4i addrspace(1)* src, v4i addrspace(3)* dst, imm int offset, imm int cpol)
#if defined(__has_builtin)
#if __has_builtin(__builtin_amdgcn_global_load_async_to_lds_b128) && \
    __has_builtin(__builtin_amdgcn_s_wait_asynccnt)
#define SF_USE_ASYNC_LDS 1
#endif
#endif
#ifndef SF_USE_ASYNC_LDS
#define SF_USE_ASYNC_LDS 0
#endif

#if SF_USE_ASYNC_LDS
typedef __attribute__((address_space(1))) v4i* sf_gptr_b128;
typedef __attribute__((address_space(3))) v4i* sf_lptr_b128;
#define SF_ASYNC_B128(gsrc, ldst) \
    __builtin_amdgcn_global_load_async_to_lds_b128( \
        (sf_gptr_b128)(gsrc), (sf_lptr_b128)(ldst), 0, 0)
#endif

// ---------------- problem constants ----------------
#define BB   512
#define TT   28
#define DD   1024
#define CC   2513
#define MM   2048          // 512 batches * 4 selected steps
#define H2   2048          // 2*H
#define INW  4096          // 4*H
#define IN4  1024          // IN/4
#define IN8  512           // IN/8

// =====================================================================
// Kernel 1: gather timesteps t = {12,16,20,24} of x0/x1, f32 -> bf16.
// Row r = b*4 + j maps to x[b, 12+4j, :].
// =====================================================================
__global__ __launch_bounds__(256) void sf_pack_acts(
    const float* __restrict__ x0, const float* __restrict__ x1,
    bf16_t* __restrict__ A0, bf16_t* __restrict__ A1)
{
    const int row = blockIdx.x;            // 0..2047
    const int b   = row >> 2;
    const int j   = row & 3;
    const int ts  = 12 + 4 * j;
    const int t   = threadIdx.x;           // 256 threads * float4 = 1024 floats

    const float4 v0 = ((const float4*)(x0 + ((size_t)b * TT + ts) * DD))[t];
    const float4 v1 = ((const float4*)(x1 + ((size_t)b * TT + ts) * DD))[t];

    bf16x4 o0 = { (bf16_t)v0.x, (bf16_t)v0.y, (bf16_t)v0.z, (bf16_t)v0.w };
    bf16x4 o1 = { (bf16_t)v1.x, (bf16_t)v1.y, (bf16_t)v1.z, (bf16_t)v1.w };

    ((bf16x4*)(A0 + (size_t)row * DD))[t] = o0;
    ((bf16x4*)(A1 + (size_t)row * DD))[t] = o1;
}

// =====================================================================
// Kernel 2: weight transpose + convert:  W (K x N) f32 -> Wt (N x K) bf16
// =====================================================================
__global__ __launch_bounds__(256) void sf_tconv(
    const float* __restrict__ W, bf16_t* __restrict__ Wt, int K, int N)
{
    __shared__ float tile[32][33];
    const int n0 = blockIdx.x * 32;
    const int k0 = blockIdx.y * 32;
    const int tx = threadIdx.x, ty = threadIdx.y;   // 32 x 8

    #pragma unroll
    for (int i = 0; i < 32; i += 8) {
        const int k = k0 + ty + i, n = n0 + tx;
        tile[ty + i][tx] = (k < K && n < N) ? W[(size_t)k * N + n] : 0.f;
    }
    __syncthreads();
    #pragma unroll
    for (int i = 0; i < 32; i += 8) {
        const int n = n0 + ty + i, k = k0 + tx;
        if (n < N && k < K) Wt[(size_t)n * K + k] = (bf16_t)tile[tx][ty + i];
    }
}

// =====================================================================
// Kernel 3: bf16 WMMA GEMM.
//   out(M x N) = act( A(M x K) * Bt(N x K)^T + bias )
//   Block = 256 threads (8 wave32).  Block tile 128x256, BK=32, LDS
//   ping-pong (one barrier per K-step).  Waves laid out 2(M) x 4(N);
//   each wave owns 64x64 = 4x4 WMMA accumulators ->
//   16 ds_load_b128 per 16 v_wmma (1.0 LDS loads / WMMA).
// =====================================================================
template<bool RELU, bool OUT_BF16>
__global__ __launch_bounds__(256) void sf_gemm_bf16(
    const bf16_t* __restrict__ A, const bf16_t* __restrict__ Bt,
    const float* __restrict__ bias, void* __restrict__ outp,
    int M, int N, int K, int ldOut)
{
    // 80-byte row stride: ds_*_b128 aligned, bank-conflict free
    __shared__ bf16_t As[2][128][40];
    __shared__ bf16_t Bs[2][256][40];

    const int tid   = threadIdx.x;
    const int lane  = tid & 31;
    const int wid   = tid >> 5;
    const int waveM = wid & 1;        // 2 wave rows  -> 64 M each
    const int waveN = wid >> 1;       // 4 wave cols  -> 64 N each
    const int half  = lane >> 4;      // lane group (0..15 / 16..31)
    const int lrow  = lane & 15;

    const int mBase = blockIdx.y * 128;
    const int nBase = blockIdx.x * 256;

    // cooperative tile-load mapping: 2 threads per 32-K row, 32B each
    const int ldr = tid >> 1;             // 0..127
    const int ldc = (tid & 1) * 16;       // 0 or 16 (bf16 elements)

    const bf16_t* aPtr = A + (size_t)(mBase + ldr) * K + ldc;
    const int  bn0 = nBase + ldr;
    const int  bn1 = nBase + 128 + ldr;
    const bool bv0 = (bn0 < N), bv1 = (bn1 < N);
    const bf16_t* bPtr0 = Bt + (size_t)(bv0 ? bn0 : 0) * K + ldc;
    const bf16_t* bPtr1 = Bt + (size_t)(bv1 ? bn1 : 0) * K + ldc;

    v8f acc[4][4];
    #pragma unroll
    for (int i = 0; i < 4; ++i)
        #pragma unroll
        for (int j = 0; j < 4; ++j)
            #pragma unroll
            for (int e = 0; e < 8; ++e) acc[i][j][e] = 0.f;

    // ---- stage one BK=32 slice of A/B tiles into LDS buffer `buf` ----
    auto stage = [&](int buf, int k0) {
#if SF_USE_ASYNC_LDS
        SF_ASYNC_B128(aPtr + k0,     &As[buf][ldr][ldc]);
        SF_ASYNC_B128(aPtr + k0 + 8, &As[buf][ldr][ldc + 8]);
        if (bv0) {
            SF_ASYNC_B128(bPtr0 + k0,     &Bs[buf][ldr][ldc]);
            SF_ASYNC_B128(bPtr0 + k0 + 8, &Bs[buf][ldr][ldc + 8]);
        }
        if (bv1) {
            SF_ASYNC_B128(bPtr1 + k0,     &Bs[buf][128 + ldr][ldc]);
            SF_ASYNC_B128(bPtr1 + k0 + 8, &Bs[buf][128 + ldr][ldc + 8]);
        }
#else
        const uint4 a0 = *(const uint4*)(aPtr + k0);
        const uint4 a1 = *(const uint4*)(aPtr + k0 + 8);
        uint4 c0 = make_uint4(0u,0u,0u,0u), c1 = c0, d0 = c0, d1 = c0;
        if (bv0) { c0 = *(const uint4*)(bPtr0 + k0); c1 = *(const uint4*)(bPtr0 + k0 + 8); }
        if (bv1) { d0 = *(const uint4*)(bPtr1 + k0); d1 = *(const uint4*)(bPtr1 + k0 + 8); }
        if (k0 + 32 < K) {
            __builtin_prefetch(aPtr + k0 + 32, 0, 3);
            __builtin_prefetch(bPtr0 + k0 + 32, 0, 3);
            __builtin_prefetch(bPtr1 + k0 + 32, 0, 3);
        }
        *(uint4*)&As[buf][ldr][ldc]           = a0;
        *(uint4*)&As[buf][ldr][ldc + 8]       = a1;
        *(uint4*)&Bs[buf][ldr][ldc]           = c0;
        *(uint4*)&Bs[buf][ldr][ldc + 8]       = c1;
        *(uint4*)&Bs[buf][128 + ldr][ldc]     = d0;
        *(uint4*)&Bs[buf][128 + ldr][ldc + 8] = d1;
#endif
    };

    stage(0, 0);
#if SF_USE_ASYNC_LDS
    __builtin_amdgcn_s_wait_asynccnt(0);
#endif
    __syncthreads();

    for (int k0 = 0; k0 < K; k0 += 32) {
        const int  buf     = (k0 >> 5) & 1;
        const bool hasNext = (k0 + 32) < K;
        if (hasNext) stage(buf ^ 1, k0 + 32);

        // B fragments (this code's contiguous k order, same as A: lanes 0-15 K=0..15, lanes 16-31 K=16..31)
        Frag bfr[4];
        #pragma unroll
        for (int nt = 0; nt < 4; ++nt) {
            const bf16_t* br = &Bs[buf][waveN * 64 + nt * 16 + lrow][0];
            bfr[nt].h[0] = *(const bf16x8*)(br + half * 16);
            bfr[nt].h[1] = *(const bf16x8*)(br + half * 16 + 8);
        }
        // A fragments (lanes 0-15: K 0..15; lanes 16-31: K 16..31)
        #pragma unroll
        for (int mt = 0; mt < 4; ++mt) {
            Frag af;
            const bf16_t* ar = &As[buf][waveM * 64 + mt * 16 + lrow][0];
            af.h[0] = *(const bf16x8*)(ar + half * 16);
            af.h[1] = *(const bf16x8*)(ar + half * 16 + 8);
            #pragma unroll
            for (int nt = 0; nt < 4; ++nt)
                acc[mt][nt] = __builtin_amdgcn_wmma_f32_16x16x32_bf16(
                    false, af.v, false, bfr[nt].v,
                    (short)0, acc[mt][nt], false, false);
        }

#if SF_USE_ASYNC_LDS
        if (hasNext) __builtin_amdgcn_s_wait_asynccnt(0);
#endif
        __syncthreads();
    }

    // ---- epilogue: bias (+ReLU); C/D layout: VGPR i -> M = i + half*8, N = lrow
    #pragma unroll
    for (int mt = 0; mt < 4; ++mt) {
        const int rBase = mBase + waveM * 64 + mt * 16 + half * 8;
        #pragma unroll
        for (int nt = 0; nt < 4; ++nt) {
            const int col = nBase + waveN * 64 + nt * 16 + lrow;
            if (col < N) {
                const float bv = bias[col];
                #pragma unroll
                for (int i = 0; i < 8; ++i) {
                    float v = acc[mt][nt][i] + bv;
                    if (RELU) v = fmaxf(v, 0.f);
                    const size_t idx = (size_t)(rBase + i) * (size_t)ldOut + col;
                    if (OUT_BF16) ((bf16_t*)outp)[idx] = (bf16_t)v;
                    else          ((float*)outp)[idx]  = v;
                }
            }
        }
    }
}

// =====================================================================
// Kernel 4: logits = h2 @ W3 + b3 (N=2), 2-way softmax, blend s0s/s1s.
// One block per output row.
// =====================================================================
__global__ __launch_bounds__(256) void sf_finalize(
    const bf16_t* __restrict__ h2, const float* __restrict__ W3,
    const float* __restrict__ b3, const float* __restrict__ s0s,
    const float* __restrict__ s1s, float* __restrict__ out)
{
    const int row = blockIdx.x;
    const int t   = threadIdx.x;

    __shared__ float r0[256], r1[256];
    __shared__ float aSh[2];

    float p0 = 0.f, p1 = 0.f;
    for (int i = t; i < IN8; i += 256) {
        const float hv = (float)h2[(size_t)row * IN8 + i];
        p0 += hv * W3[i * 2 + 0];
        p1 += hv * W3[i * 2 + 1];
    }
    r0[t] = p0; r1[t] = p1;
    __syncthreads();
    for (int s = 128; s > 0; s >>= 1) {
        if (t < s) { r0[t] += r0[t + s]; r1[t] += r1[t + s]; }
        __syncthreads();
    }
    if (t == 0) {
        const float l0 = r0[0] + b3[0];
        const float l1 = r1[0] + b3[1];
        const float m  = fmaxf(l0, l1);
        const float e0 = __expf(l0 - m);
        const float e1 = __expf(l1 - m);
        const float inv = 1.f / (e0 + e1);
        aSh[0] = e0 * inv; aSh[1] = e1 * inv;
    }
    __syncthreads();
    const float a0 = aSh[0], a1 = aSh[1];
    const size_t base = (size_t)row * CC;
    for (int c = t; c < CC; c += 256)
        out[base + c] = s0s[base + c] * a0 + s1s[base + c] * a1;
}

// =====================================================================
// Host-side orchestration
// =====================================================================
extern "C" void kernel_launch(void* const* d_in, const int* in_sizes, int n_in,
                              void* d_out, int out_size, void* d_ws, size_t ws_size,
                              hipStream_t stream)
{
    (void)in_sizes; (void)n_in; (void)out_size; (void)ws_size;

    const float* x0  = (const float*)d_in[0];
    const float* x1  = (const float*)d_in[1];
    const float* Ws0 = (const float*)d_in[2];
    const float* bs0 = (const float*)d_in[3];
    const float* Wc0 = (const float*)d_in[4];
    const float* bc0 = (const float*)d_in[5];
    const float* Ws1 = (const float*)d_in[6];
    const float* bs1 = (const float*)d_in[7];
    const float* Wc1 = (const float*)d_in[8];
    const float* bc1 = (const float*)d_in[9];
    const float* W1  = (const float*)d_in[10];
    const float* b1  = (const float*)d_in[11];
    const float* W2  = (const float*)d_in[12];
    const float* b2  = (const float*)d_in[13];
    const float* W3  = (const float*)d_in[14];
    const float* b3  = (const float*)d_in[15];
    float* out = (float*)d_out;

    // ---- workspace carve-up (256B aligned) ----
    char*  ws  = (char*)d_ws;
    size_t off = 0;
    auto carve = [&](size_t bytes) -> void* {
        void* p = ws + off;
        off += (bytes + 255) & ~(size_t)255;
        return p;
    };
    bf16_t* A0   = (bf16_t*)carve((size_t)MM * DD * 2);
    bf16_t* A1   = (bf16_t*)carve((size_t)MM * DD * 2);
    bf16_t* Ws0t = (bf16_t*)carve((size_t)CC * DD * 2);
    bf16_t* Ws1t = (bf16_t*)carve((size_t)CC * DD * 2);
    bf16_t* Wc0t = (bf16_t*)carve((size_t)H2 * DD * 2);
    bf16_t* Wc1t = (bf16_t*)carve((size_t)H2 * DD * 2);
    bf16_t* W1t  = (bf16_t*)carve((size_t)IN4 * INW * 2);
    bf16_t* W2t  = (bf16_t*)carve((size_t)IN8 * IN4 * 2);
    float*  s0s  = (float*)carve((size_t)MM * CC * 4);
    float*  s1s  = (float*)carve((size_t)MM * CC * 4);
    bf16_t* ctx  = (bf16_t*)carve((size_t)MM * INW * 2);
    bf16_t* h1   = (bf16_t*)carve((size_t)MM * IN4 * 2);
    bf16_t* h2   = (bf16_t*)carve((size_t)MM * IN8 * 2);

    const dim3 blk256(256, 1, 1);

    // 1) gather + convert activations
    sf_pack_acts<<<dim3(MM, 1, 1), blk256, 0, stream>>>(x0, x1, A0, A1);

    // 2) weight transpose + convert
    const dim3 tb(32, 8, 1);
    sf_tconv<<<dim3((CC + 31) / 32, DD / 32, 1),  tb, 0, stream>>>(Ws0, Ws0t, DD, CC);
    sf_tconv<<<dim3((CC + 31) / 32, DD / 32, 1),  tb, 0, stream>>>(Ws1, Ws1t, DD, CC);
    sf_tconv<<<dim3(H2 / 32, DD / 32, 1),         tb, 0, stream>>>(Wc0, Wc0t, DD, H2);
    sf_tconv<<<dim3(H2 / 32, DD / 32, 1),         tb, 0, stream>>>(Wc1, Wc1t, DD, H2);
    sf_tconv<<<dim3(IN4 / 32, INW / 32, 1),       tb, 0, stream>>>(W1, W1t, INW, IN4);
    sf_tconv<<<dim3(IN8 / 32, IN4 / 32, 1),       tb, 0, stream>>>(W2, W2t, IN4, IN8);

    // 3) WMMA GEMMs (block tile 128 x 256)
    const int mg = MM / 128;   // 16
    // s-paths: f32 output
    sf_gemm_bf16<false, false><<<dim3((CC + 255) / 256, mg, 1), blk256, 0, stream>>>(
        A0, Ws0t, bs0, (void*)s0s, MM, CC, DD, CC);
    sf_gemm_bf16<false, false><<<dim3((CC + 255) / 256, mg, 1), blk256, 0, stream>>>(
        A1, Ws1t, bs1, (void*)s1s, MM, CC, DD, CC);
    // c-paths: bf16 output straight into ctx (col 0 and col 2048)
    sf_gemm_bf16<false, true><<<dim3(H2 / 256, mg, 1), blk256, 0, stream>>>(
        A0, Wc0t, bc0, (void*)ctx, MM, H2, DD, INW);
    sf_gemm_bf16<false, true><<<dim3(H2 / 256, mg, 1), blk256, 0, stream>>>(
        A1, Wc1t, bc1, (void*)(ctx + H2), MM, H2, DD, INW);
    // MLP
    sf_gemm_bf16<true, true><<<dim3(IN4 / 256, mg, 1), blk256, 0, stream>>>(
        ctx, W1t, b1, (void*)h1, MM, IN4, INW, IN4);
    sf_gemm_bf16<true, true><<<dim3(IN8 / 256, mg, 1), blk256, 0, stream>>>(
        h1, W2t, b2, (void*)h2, MM, IN8, IN4, IN8);

    // 4) softmax gate + blend
    sf_finalize<<<dim3(MM, 1, 1), blk256, 0, stream>>>(h2, W3, b3, s0s, s1s, out);
}
